// TRACKLeftModel_21354577395791
// MI455X (gfx1250) — matrix operand order repaired and checked by reading, not compiled, not run on hardware
//
#include <hip/hip_runtime.h>
#include <hip/hip_bf16.h>

typedef __bf16 bf16_t;
typedef __bf16 v16bf __attribute__((ext_vector_type(16)));
typedef __bf16 v8bf  __attribute__((ext_vector_type(8)));
typedef float  v8f   __attribute__((ext_vector_type(8)));
typedef float  v8fv  __attribute__((ext_vector_type(8)));

#define HID   256
#define GATE4 1024
#define BATCH 1024
#define MENC  5
#define HWIN  25
#define SALK  8192

#define WMMA_BF16(a, b, c) \
  __builtin_amdgcn_wmma_f32_16x16x32_bf16(false, (a), false, (b), (short)0, (c), false, false)

__device__ __forceinline__ float sigm(float x){ return 1.0f / (1.0f + __expf(-x)); }

// float8 x2 -> bf16 x16 via paired v_cvt_pk_bf16_f32 (no half-moves)
__device__ __forceinline__ v16bf cvt_a(v8fv lo, v8fv hi){
  v8bf l = __builtin_convertvector(lo, v8bf);
  v8bf h = __builtin_convertvector(hi, v8bf);
  return __builtin_shufflevector(l, h, 0,1,2,3,4,5,6,7,8,9,10,11,12,13,14,15);
}

// Map (row 0..15, k) -> half-index inside packed WMMA-A bf16 layout.
// Per 16x32 k-tile: lane l holds row (l&15); halves m<8 -> k_off=(l>>4)*8+m,
// halves m>=8 -> k_off=16+(l>>4)*8+(m-8). Stored as ((kt*32 + lane)*16 + m).
__device__ __forceinline__ int packA_off(int r, int k){
  int kt = k >> 5;
  int ko = k & 31;
  int hi, m;
  if (ko < 16) { hi = ko >> 3; m = ko & 7; }
  else         { int z = ko - 16; hi = z >> 3; m = 8 + (z & 7); }
  return ((kt << 5) + (hi << 4) + r) * 16 + m;
}

// ---------------------------------------------------------------------------
// Pack fp32 weight W[rowOff+k][n] (ld srcN) into bf16 WMMA-B layout.
// ---------------------------------------------------------------------------
__global__ void pack_b_kernel(const float* __restrict__ W, int srcN, int rowOff,
                              int K, int N, bf16_t* __restrict__ out){
  int idx = blockIdx.x * 256 + threadIdx.x;
  if (idx >= K * N) return;
  int m    = idx & 15;
  int l    = (idx >> 4) & 31;
  int tile = idx >> 9;
  int nkt  = K >> 5;
  int kT   = tile % nkt;
  int nT   = tile / nkt;
  int k = (kT << 5) + ((m < 8) ? (((l >> 4) << 3) + m) : (16 + ((l >> 4) << 3) + (m - 8)));
  int n = (nT << 4) + (l & 15);
  out[idx] = (bf16_t)W[(size_t)(rowOff + k) * srcN + n];
}

// ---------------------------------------------------------------------------
// Fuse position path:  Wf[q][n] = sum_k Wp[q][k]*Wih[rowOff+k][n];  bf likewise.
// ---------------------------------------------------------------------------
__global__ void fuse_pos_kernel(const float* __restrict__ Wp,   // 3 x HID
                                const float* __restrict__ bp,   // HID
                                const float* __restrict__ Wih,  // (2H) x GATE4
                                int rowOff,
                                float* __restrict__ Wf,         // 3 x GATE4
                                float* __restrict__ bf){        // GATE4
  int n = blockIdx.x * 256 + threadIdx.x;
  if (n >= GATE4) return;
  float a0 = 0.f, a1 = 0.f, a2 = 0.f, ab = 0.f;
  for (int k = 0; k < HID; ++k){
    float w = Wih[(size_t)(rowOff + k) * GATE4 + n];
    a0 += Wp[k]           * w;
    a1 += Wp[HID + k]     * w;
    a2 += Wp[2 * HID + k] * w;
    ab += bp[k]           * w;
  }
  Wf[n] = a0; Wf[GATE4 + n] = a1; Wf[2 * GATE4 + n] = a2; bf[n] = ab;
}

// ---------------------------------------------------------------------------
// G1 (HBM-bound stream): out = A(fp32 RxK) @ Bp(bf16 packed Kx256) + bias,
// emitted as packed-A bf16 (Rx256).  Ping-pong double buffer, unrolled x2:
// even stage consumes buf0 while loading buf1, odd stage the reverse.
// ---------------------------------------------------------------------------
__global__ __launch_bounds__(256)
void sal_gemm_kernel(const float* __restrict__ A, int K,
                     const bf16_t* __restrict__ Bp,
                     const float* __restrict__ bias,
                     bf16_t* __restrict__ outP){
  __shared__ __align__(32) float tile[32 * 256];
  const int lane = threadIdx.x & 31;
  const int wave = threadIdx.x >> 5;
  const int lrt  = wave >> 2;
  const int colBase = (wave & 3) * 64;
  const int hi = lane >> 4;
  const int ln = lane & 15;
  const int rowTile = blockIdx.x * 2 + lrt;
  const int nkt = K >> 5;   // always even here (256)
  const float* __restrict__ aptr = A + (size_t)(rowTile * 16 + ln) * K + (hi << 3);

  const bf16_t* __restrict__ bptr0 = Bp + ((size_t)((colBase >> 4) + 0) * nkt * 32 + lane) * 16;
  const bf16_t* __restrict__ bptr1 = Bp + ((size_t)((colBase >> 4) + 1) * nkt * 32 + lane) * 16;
  const bf16_t* __restrict__ bptr2 = Bp + ((size_t)((colBase >> 4) + 2) * nkt * 32 + lane) * 16;
  const bf16_t* __restrict__ bptr3 = Bp + ((size_t)((colBase >> 4) + 3) * nkt * 32 + lane) * 16;

  v8f acc[4];
#pragma unroll
  for (int t = 0; t < 4; ++t){
    float b = bias[colBase + t * 16 + ln];
#pragma unroll
    for (int v = 0; v < 8; ++v) acc[t][v] = b;
  }

  // prologue: kt = 0 into buffer 0
  v8fv fl0 = *(const v8fv*)(aptr);
  v8fv fh0 = *(const v8fv*)(aptr + 16);
  v16bf b00 = *(const v16bf*)(bptr0);
  v16bf b01 = *(const v16bf*)(bptr1);
  v16bf b02 = *(const v16bf*)(bptr2);
  v16bf b03 = *(const v16bf*)(bptr3);
  v8fv fl1, fh1;
  v16bf b10, b11, b12, b13;

  for (int kt = 0; kt < nkt; kt += 2){
    // load kt+1 into buffer 1, consume buffer 0
    {
      const float* ap = aptr + ((kt + 1) << 5);
      __builtin_prefetch(ap + 1024, 0, 0);
      fl1 = *(const v8fv*)(ap);
      fh1 = *(const v8fv*)(ap + 16);
      size_t bo = (size_t)(kt + 1) * 512;
      b10 = *(const v16bf*)(bptr0 + bo);
      b11 = *(const v16bf*)(bptr1 + bo);
      b12 = *(const v16bf*)(bptr2 + bo);
      b13 = *(const v16bf*)(bptr3 + bo);
    }
    v16bf a0 = cvt_a(fl0, fh0);
    acc[0] = WMMA_BF16(a0, b00, acc[0]);
    acc[1] = WMMA_BF16(a0, b01, acc[1]);
    acc[2] = WMMA_BF16(a0, b02, acc[2]);
    acc[3] = WMMA_BF16(a0, b03, acc[3]);

    // load kt+2 into buffer 0, consume buffer 1
    if (kt + 2 < nkt){
      const float* ap = aptr + ((kt + 2) << 5);
      __builtin_prefetch(ap + 1024, 0, 0);
      fl0 = *(const v8fv*)(ap);
      fh0 = *(const v8fv*)(ap + 16);
      size_t bo = (size_t)(kt + 2) * 512;
      b00 = *(const v16bf*)(bptr0 + bo);
      b01 = *(const v16bf*)(bptr1 + bo);
      b02 = *(const v16bf*)(bptr2 + bo);
      b03 = *(const v16bf*)(bptr3 + bo);
    }
    v16bf a1 = cvt_a(fl1, fh1);
    acc[0] = WMMA_BF16(a1, b10, acc[0]);
    acc[1] = WMMA_BF16(a1, b11, acc[1]);
    acc[2] = WMMA_BF16(a1, b12, acc[2]);
    acc[3] = WMMA_BF16(a1, b13, acc[3]);
  }

#pragma unroll
  for (int t = 0; t < 4; ++t)
#pragma unroll
    for (int v = 0; v < 8; ++v)
      tile[(lrt * 16 + v + 8 * hi) * 256 + colBase + t * 16 + ln] = acc[t][v];
  __syncthreads();

  for (int u = 0; u < 32; ++u){
    int p    = threadIdx.x + u * 256;
    int m    = p & 15;
    int l    = (p >> 4) & 31;
    int tI   = p >> 9;
    int plrt = tI >> 3;
    int lkt  = tI & 7;
    int rloc = plrt * 16 + (l & 15);
    int koff = (m < 8) ? (((l >> 4) << 3) + m) : (16 + ((l >> 4) << 3) + (m - 8));
    int kloc = (lkt << 5) + koff;
    int drt  = blockIdx.x * 2 + plrt;
    outP[((size_t)(drt * 8 + lkt) * 32 + l) * 16 + m] = (bf16_t)tile[rloc * 256 + kloc];
  }
}

// ---------------------------------------------------------------------------
// G2: out(fp32 RxN) = Ap(packed bf16) @ Bp(packed bf16 KxN)
//     + bias1 + bias2 [+ bias3] [+ posv(row,3) @ posW(3,N)].  Ping-pong x2.
// ---------------------------------------------------------------------------
__global__ __launch_bounds__(256)
void gemm_packA_kernel(const bf16_t* __restrict__ Ap, int ankt,
                       const bf16_t* __restrict__ Bp,
                       const float* __restrict__ bias1,
                       const float* __restrict__ bias2,
                       const float* __restrict__ bias3,
                       const float* __restrict__ posv,
                       const float* __restrict__ posW,
                       float* __restrict__ out, int N){
  const int lane = threadIdx.x & 31;
  const int wave = threadIdx.x >> 5;
  const int lrt  = wave >> 2;
  const int colBase = blockIdx.y * 256 + (wave & 3) * 64;
  const int hi = lane >> 4;
  const int ln = lane & 15;
  const int rowTile = blockIdx.x * 2 + lrt;

  v8f acc[4];
#pragma unroll
  for (int t = 0; t < 4; ++t){
    int col = colBase + t * 16 + ln;
    float b = bias1[col] + bias2[col];
    if (bias3) b += bias3[col];
#pragma unroll
    for (int v = 0; v < 8; ++v){
      float val = b;
      if (posW){
        int row = rowTile * 16 + v + 8 * hi;
        val += posv[row * 3 + 0] * posW[col]
             + posv[row * 3 + 1] * posW[N + col]
             + posv[row * 3 + 2] * posW[2 * N + col];
      }
      acc[t][v] = val;
    }
  }

  const bf16_t* __restrict__ apB   = Ap + (size_t)rowTile * ankt * 512 + (size_t)lane * 16;
  const bf16_t* __restrict__ bptr0 = Bp + ((size_t)((colBase >> 4) + 0) * ankt * 32 + lane) * 16;
  const bf16_t* __restrict__ bptr1 = Bp + ((size_t)((colBase >> 4) + 1) * ankt * 32 + lane) * 16;
  const bf16_t* __restrict__ bptr2 = Bp + ((size_t)((colBase >> 4) + 2) * ankt * 32 + lane) * 16;
  const bf16_t* __restrict__ bptr3 = Bp + ((size_t)((colBase >> 4) + 3) * ankt * 32 + lane) * 16;

  v16bf a0  = *(const v16bf*)(apB);
  v16bf b00 = *(const v16bf*)(bptr0);
  v16bf b01 = *(const v16bf*)(bptr1);
  v16bf b02 = *(const v16bf*)(bptr2);
  v16bf b03 = *(const v16bf*)(bptr3);
  v16bf a1, b10, b11, b12, b13;

  for (int kt = 0; kt < ankt; kt += 2){   // ankt even (8)
    {
      size_t bo = (size_t)(kt + 1) * 512;
      a1  = *(const v16bf*)(apB + bo);
      b10 = *(const v16bf*)(bptr0 + bo);
      b11 = *(const v16bf*)(bptr1 + bo);
      b12 = *(const v16bf*)(bptr2 + bo);
      b13 = *(const v16bf*)(bptr3 + bo);
    }
    acc[0] = WMMA_BF16(a0, b00, acc[0]);
    acc[1] = WMMA_BF16(a0, b01, acc[1]);
    acc[2] = WMMA_BF16(a0, b02, acc[2]);
    acc[3] = WMMA_BF16(a0, b03, acc[3]);
    if (kt + 2 < ankt){
      size_t bo = (size_t)(kt + 2) * 512;
      a0  = *(const v16bf*)(apB + bo);
      b00 = *(const v16bf*)(bptr0 + bo);
      b01 = *(const v16bf*)(bptr1 + bo);
      b02 = *(const v16bf*)(bptr2 + bo);
      b03 = *(const v16bf*)(bptr3 + bo);
    }
    acc[0] = WMMA_BF16(a1, b10, acc[0]);
    acc[1] = WMMA_BF16(a1, b11, acc[1]);
    acc[2] = WMMA_BF16(a1, b12, acc[2]);
    acc[3] = WMMA_BF16(a1, b13, acc[3]);
  }

#pragma unroll
  for (int t = 0; t < 4; ++t){
    int col = colBase + t * 16 + ln;
#pragma unroll
    for (int v = 0; v < 8; ++v)
      out[(size_t)(rowTile * 16 + v + 8 * hi) * N + col] = acc[t][v];
  }
}

// ---------------------------------------------------------------------------
// Encoder LSTM: block owns 16 batch rows for all MENC steps; h packed in LDS.
// ---------------------------------------------------------------------------
__global__ __launch_bounds__(256)
void lstm_enc_kernel(const float* __restrict__ gpre,   // (B*MENC) x 1024
                     const bf16_t* __restrict__ WhhP,  // packed 256x1024
                     float* __restrict__ hF, float* __restrict__ cF){
  __shared__ __align__(32) bf16_t hP[8 * 32 * 16];
  __shared__ float cS[16 * HID];
  __shared__ float gS[16 * 512];
  const int tid  = threadIdx.x;
  const int lane = tid & 31, wave = tid >> 5;
  const int hi = lane >> 4, ln = lane & 15;
  const int b0 = blockIdx.x * 16;
  const int gt = wave >> 1, half = wave & 1;

  for (int u = 0; u < 16; ++u){
    int gi = tid + u * 256;
    cS[gi] = 0.f;
    hP[packA_off(gi >> 8, gi & 255)] = (bf16_t)0.f;
  }
  __syncthreads();

  for (int t = 0; t < MENC; ++t){
    v16bf aF[8];
#pragma unroll
    for (int kt = 0; kt < 8; ++kt)
      aF[kt] = *(const v16bf*)(hP + (kt * 32 + lane) * 16);

    float hreg[16];
#pragma unroll
    for (int p = 0; p < 2; ++p){
#pragma unroll
      for (int tt = 0; tt < 4; ++tt){
        int nT  = gt * 16 + p * 8 + half * 4 + tt;
        int col = nT * 16 + ln;
        v8f acc;
#pragma unroll
        for (int v = 0; v < 8; ++v)
          acc[v] = gpre[(size_t)((b0 + v + 8 * hi) * MENC + t) * GATE4 + col];
        const bf16_t* __restrict__ bbase = WhhP + ((size_t)nT * 8 * 32 + lane) * 16;
        v16bf bcur = *(const v16bf*)(bbase);
#pragma unroll
        for (int kt = 0; kt < 8; ++kt){
          v16bf bnext;
          if (kt < 7) bnext = *(const v16bf*)(bbase + (size_t)(kt + 1) * 512);
          acc = WMMA_BF16(aF[kt], bcur, acc);
          bcur = bnext;
        }
        int cl = half * 64 + tt * 16 + ln;
#pragma unroll
        for (int v = 0; v < 8; ++v)
          gS[(v + 8 * hi) * 512 + gt * 128 + cl] = acc[v];
      }
      __syncthreads();
#pragma unroll
      for (int it = 0; it < 8; ++it){
        int gi = tid + it * 256;
        int r = gi >> 7, jl = gi & 127;
        int j = p * 128 + jl;
        float ig = gS[r * 512 + jl];
        float fg = gS[r * 512 + 128 + jl];
        float gg = gS[r * 512 + 256 + jl];
        float og = gS[r * 512 + 384 + jl];
        float c  = sigm(fg) * cS[r * HID + j] + sigm(ig) * tanhf(gg);
        float h  = sigm(og) * tanhf(c);
        cS[r * HID + j] = c;
        hreg[p * 8 + it] = h;
        if (t == MENC - 1) hF[(size_t)(b0 + r) * HID + j] = h;
      }
      __syncthreads();
    }
#pragma unroll
    for (int p = 0; p < 2; ++p)
#pragma unroll
      for (int it = 0; it < 8; ++it){
        int gi = tid + it * 256;
        hP[packA_off(gi >> 7, p * 128 + (gi & 127))] = (bf16_t)hreg[p * 8 + it];
      }
    __syncthreads();
  }
  for (int u = 0; u < 16; ++u){
    int gi = tid + u * 256;
    cF[(size_t)b0 * HID + gi] = cS[gi];
  }
}

// ---------------------------------------------------------------------------
// Decoder LSTM + head.
// ---------------------------------------------------------------------------
__global__ __launch_bounds__(256)
void lstm_dec_kernel(const float* __restrict__ h0, const float* __restrict__ c0,
                     const float* __restrict__ gpre,   // (B*HWIN) x 1024
                     const bf16_t* __restrict__ WhhP,  // packed 256x1024
                     const float* __restrict__ Wfd,    // 3 x 1024 (fused pos path)
                     const bf16_t* __restrict__ W2P,   // packed 256x256
                     const float* __restrict__ b2,
                     const float* __restrict__ Wo,     // 256 x 3
                     const float* __restrict__ bo,     // 3
                     const float* __restrict__ dpos,   // B x 3
                     float* __restrict__ out){         // B x HWIN x 3
  __shared__ __align__(32) bf16_t hP[8 * 32 * 16];
  __shared__ float cS[16 * HID];
  __shared__ float gS[16 * 512];
  __shared__ float posS[16 * 3];
  __shared__ float dS[16 * 3];
  const int tid  = threadIdx.x;
  const int lane = tid & 31, wave = tid >> 5;
  const int hi = lane >> 4, ln = lane & 15;
  const int b0 = blockIdx.x * 16;
  const int gt = wave >> 1, half = wave & 1;

  for (int u = 0; u < 16; ++u){
    int gi = tid + u * 256;
    cS[gi] = c0[(size_t)b0 * HID + gi];
    hP[packA_off(gi >> 8, gi & 255)] = (bf16_t)h0[(size_t)b0 * HID + gi];
  }
  if (tid < 48) posS[tid] = dpos[(size_t)b0 * 3 + tid];
  __syncthreads();

  for (int t = 0; t < HWIN; ++t){
    v16bf aF[8];
#pragma unroll
    for (int kt = 0; kt < 8; ++kt)
      aF[kt] = *(const v16bf*)(hP + (kt * 32 + lane) * 16);

    float hreg[16];
#pragma unroll
    for (int p = 0; p < 2; ++p){
#pragma unroll
      for (int tt = 0; tt < 4; ++tt){
        int nT  = gt * 16 + p * 8 + half * 4 + tt;
        int col = nT * 16 + ln;
        float w0 = Wfd[col], w1 = Wfd[GATE4 + col], w2 = Wfd[2 * GATE4 + col];
        v8f acc;
#pragma unroll
        for (int v = 0; v < 8; ++v){
          int r = v + 8 * hi;
          acc[v] = gpre[(size_t)((b0 + r) * HWIN + t) * GATE4 + col]
                 + posS[r * 3 + 0] * w0 + posS[r * 3 + 1] * w1 + posS[r * 3 + 2] * w2;
        }
        const bf16_t* __restrict__ bbase = WhhP + ((size_t)nT * 8 * 32 + lane) * 16;
        v16bf bcur = *(const v16bf*)(bbase);
#pragma unroll
        for (int kt = 0; kt < 8; ++kt){
          v16bf bnext;
          if (kt < 7) bnext = *(const v16bf*)(bbase + (size_t)(kt + 1) * 512);
          acc = WMMA_BF16(aF[kt], bcur, acc);
          bcur = bnext;
        }
        int cl = half * 64 + tt * 16 + ln;
#pragma unroll
        for (int v = 0; v < 8; ++v)
          gS[(v + 8 * hi) * 512 + gt * 128 + cl] = acc[v];
      }
      __syncthreads();
#pragma unroll
      for (int it = 0; it < 8; ++it){
        int gi = tid + it * 256;
        int r = gi >> 7, jl = gi & 127;
        int j = p * 128 + jl;
        float ig = gS[r * 512 + jl];
        float fg = gS[r * 512 + 128 + jl];
        float gg = gS[r * 512 + 256 + jl];
        float og = gS[r * 512 + 384 + jl];
        float c  = sigm(fg) * cS[r * HID + j] + sigm(ig) * tanhf(gg);
        float h  = sigm(og) * tanhf(c);
        cS[r * HID + j] = c;
        hreg[p * 8 + it] = h;
      }
      __syncthreads();
    }
#pragma unroll
    for (int p = 0; p < 2; ++p)
#pragma unroll
      for (int it = 0; it < 8; ++it){
        int gi = tid + it * 256;
        hP[packA_off(gi >> 7, p * 128 + (gi & 127))] = (bf16_t)hreg[p * 8 + it];
      }
    __syncthreads();

    // t2 = h @ W2 + b2 -> gS[0:16*256]
    v16bf aG[8];
#pragma unroll
    for (int kt = 0; kt < 8; ++kt)
      aG[kt] = *(const v16bf*)(hP + (kt * 32 + lane) * 16);
#pragma unroll
    for (int tt = 0; tt < 2; ++tt){
      int nT  = wave * 2 + tt;
      int col = nT * 16 + ln;
      v8f acc;
      float b = b2[col];
#pragma unroll
      for (int v = 0; v < 8; ++v) acc[v] = b;
      const bf16_t* __restrict__ bbase = W2P + ((size_t)nT * 8 * 32 + lane) * 16;
      v16bf bcur = *(const v16bf*)(bbase);
#pragma unroll
      for (int kt = 0; kt < 8; ++kt){
        v16bf bnext;
        if (kt < 7) bnext = *(const v16bf*)(bbase + (size_t)(kt + 1) * 512);
        acc = WMMA_BF16(aG[kt], bcur, acc);
        bcur = bnext;
      }
#pragma unroll
      for (int v = 0; v < 8; ++v)
        gS[(v + 8 * hi) * 256 + col] = acc[v];
    }
    __syncthreads();

    if (tid < 48){
      int r = tid / 3, jo = tid % 3;
      float d = bo[jo];
      for (int k = 0; k < HID; ++k) d += gS[r * 256 + k] * Wo[k * 3 + jo];
      dS[r * 3 + jo] = d;
    }
    __syncthreads();

    if (tid < 16){
      int r = tid;
      float p0 = posS[r * 3 + 0] + dS[r * 3 + 0];
      float p1 = posS[r * 3 + 1] + dS[r * 3 + 1];
      float p2 = posS[r * 3 + 2] + dS[r * 3 + 2];
      float inv = rsqrtf(p0 * p0 + p1 * p1 + p2 * p2);
      p0 *= inv; p1 *= inv; p2 *= inv;
      posS[r * 3 + 0] = p0; posS[r * 3 + 1] = p1; posS[r * 3 + 2] = p2;
      size_t ob = ((size_t)(b0 + r) * HWIN + t) * 3;
      out[ob] = p0; out[ob + 1] = p1; out[ob + 2] = p2;
    }
    __syncthreads();
  }
}

// ---------------------------------------------------------------------------
extern "C" void kernel_launch(void* const* d_in, const int* in_sizes, int n_in,
                              void* d_out, int out_size, void* d_ws, size_t ws_size,
                              hipStream_t stream){
  const float* enc_pos = (const float*)d_in[0];
  const float* enc_sal = (const float*)d_in[1];
  const float* dec_pos = (const float*)d_in[2];
  const float* dec_sal = (const float*)d_in[3];
  const float* Wpe = (const float*)d_in[4];
  const float* bpe = (const float*)d_in[5];
  const float* Wse = (const float*)d_in[6];
  const float* bse = (const float*)d_in[7];
  const float* Wih_e = (const float*)d_in[8];
  const float* Whh_e = (const float*)d_in[9];
  const float* bih_e = (const float*)d_in[10];
  const float* bhh_e = (const float*)d_in[11];
  const float* Wih_d = (const float*)d_in[12];
  const float* Whh_d = (const float*)d_in[13];
  const float* bih_d = (const float*)d_in[14];
  const float* bhh_d = (const float*)d_in[15];
  const float* Wpd = (const float*)d_in[16];
  const float* bpd = (const float*)d_in[17];
  const float* Wsd = (const float*)d_in[18];
  const float* bsd = (const float*)d_in[19];
  const float* W2  = (const float*)d_in[20];
  const float* b2  = (const float*)d_in[21];
  const float* Wo  = (const float*)d_in[22];
  const float* bo  = (const float*)d_in[23];

  char* ws = (char*)d_ws;
  size_t off = 0;
  auto carve = [&](size_t bytes) -> char* {
    char* p = ws + off;
    off += (bytes + 255) & ~(size_t)255;
    return p;
  };
  bf16_t* WseP   = (bf16_t*)carve((size_t)SALK * 256 * 2);
  bf16_t* WsdP   = (bf16_t*)carve((size_t)SALK * 256 * 2);
  bf16_t* WihE1P = (bf16_t*)carve((size_t)256 * 1024 * 2);
  bf16_t* WhhEP  = (bf16_t*)carve((size_t)256 * 1024 * 2);
  bf16_t* WihD1P = (bf16_t*)carve((size_t)256 * 1024 * 2);
  bf16_t* WhhDP  = (bf16_t*)carve((size_t)256 * 1024 * 2);
  bf16_t* W2P    = (bf16_t*)carve((size_t)256 * 256 * 2);
  float*  WfE    = (float*)carve((size_t)3 * 1024 * 4);
  float*  bfE    = (float*)carve((size_t)1024 * 4);
  float*  WfD    = (float*)carve((size_t)3 * 1024 * 4);
  float*  bfD    = (float*)carve((size_t)1024 * 4);
  bf16_t* salEP  = (bf16_t*)carve((size_t)BATCH * MENC * 256 * 2);
  bf16_t* salDP  = (bf16_t*)carve((size_t)BATCH * HWIN * 256 * 2);
  float*  gpreE  = (float*)carve((size_t)BATCH * MENC * GATE4 * 4);
  float*  gpreD  = (float*)carve((size_t)BATCH * HWIN * GATE4 * 4);
  float*  hF     = (float*)carve((size_t)BATCH * HID * 4);
  float*  cF     = (float*)carve((size_t)BATCH * HID * 4);

  pack_b_kernel<<<(SALK * 256) / 256, 256, 0, stream>>>(Wse, 256, 0, SALK, 256, WseP);
  pack_b_kernel<<<(SALK * 256) / 256, 256, 0, stream>>>(Wsd, 256, 0, SALK, 256, WsdP);
  pack_b_kernel<<<(256 * 1024) / 256, 256, 0, stream>>>(Wih_e, 1024, 0, 256, 1024, WihE1P);
  pack_b_kernel<<<(256 * 1024) / 256, 256, 0, stream>>>(Whh_e, 1024, 0, 256, 1024, WhhEP);
  pack_b_kernel<<<(256 * 1024) / 256, 256, 0, stream>>>(Wih_d, 1024, 0, 256, 1024, WihD1P);
  pack_b_kernel<<<(256 * 1024) / 256, 256, 0, stream>>>(Whh_d, 1024, 0, 256, 1024, WhhDP);
  pack_b_kernel<<<(256 * 256) / 256, 256, 0, stream>>>(W2, 256, 0, 256, 256, W2P);
  fuse_pos_kernel<<<4, 256, 0, stream>>>(Wpe, bpe, Wih_e, 256, WfE, bfE);
  fuse_pos_kernel<<<4, 256, 0, stream>>>(Wpd, bpd, Wih_d, 256, WfD, bfD);

  sal_gemm_kernel<<<(BATCH * MENC) / 32, 256, 0, stream>>>(enc_sal, SALK, WseP, bse, salEP);
  sal_gemm_kernel<<<(BATCH * HWIN) / 32, 256, 0, stream>>>(dec_sal, SALK, WsdP, bsd, salDP);

  gemm_packA_kernel<<<dim3((BATCH * MENC) / 32, 4), 256, 0, stream>>>(
      salEP, 8, WihE1P, bih_e, bhh_e, bfE, enc_pos, WfE, gpreE, GATE4);
  gemm_packA_kernel<<<dim3((BATCH * HWIN) / 32, 4), 256, 0, stream>>>(
      salDP, 8, WihD1P, bih_d, bhh_d, bfD, nullptr, nullptr, gpreD, GATE4);

  lstm_enc_kernel<<<BATCH / 16, 256, 0, stream>>>(gpreE, WhhEP, hF, cF);
  lstm_dec_kernel<<<BATCH / 16, 256, 0, stream>>>(hF, cF, gpreD, WhhDP, WfD, W2P,
                                                  b2, Wo, bo, dec_pos, (float*)d_out);
}
